// MultiHeadAttention_50233937494502
// MI455X (gfx1250) — compile-verified, matrix operand order checked by reading, not run on hardware
//
#include <hip/hip_runtime.h>
#include <hip/hip_bf16.h>

typedef __attribute__((ext_vector_type(16))) _Float16 v16h;
typedef __attribute__((ext_vector_type(8)))  float    v8f;
typedef __attribute__((ext_vector_type(4)))  int      v4i;

#define S_LEN 4096
#define EMB   1024
#define NHEAD 16
#define HDIM  64

#if __has_builtin(__builtin_amdgcn_global_load_async_to_lds_b128)
#define HAVE_ASYNC_LDS 1
typedef __attribute__((address_space(1))) v4i* gptr_v4i;
typedef __attribute__((address_space(3))) v4i* lptr_v4i;
#else
#define HAVE_ASYNC_LDS 0
#endif

__device__ inline void wait_asynccnt0() {
#if __has_builtin(__builtin_amdgcn_s_wait_asynccnt)
  __builtin_amdgcn_s_wait_asynccnt(0);
#else
  asm volatile("s_wait_asynccnt 0x0" ::: "memory");
#endif
}

// ---------------------------------------------------------------------------
// WMMA fragment loaders (gfx1250 wave32 layouts, CDNA5 ISA §7.12.2).
// A 16x32 f16: lane holds row m=lane&15; kb=(lane>>4)*8; halves j<8 -> k=kb+j,
// j>=8 -> k=16+kb+(j-8). Two contiguous 8-half runs -> compiler emits b128.
// frag_kmaj: same per-lane shape; row index = lane&15 indexes the slow
// dimension of a K-contiguous ([row][k]) source. Used for A tiles, B tiles
// staged transposed in LDS, and K/V stored K-contiguous.
// C/D 16x16 f32: lane n=lane&15, reg r -> row m = r + (lane>>4)*8.
// ---------------------------------------------------------------------------
__device__ inline v16h frag_kmaj(const _Float16* __restrict__ src, int ld) {
  const int lane = threadIdx.x & 31;
  const int m  = lane & 15;
  const int kb = (lane >> 4) << 3;
  v16h a;
#pragma unroll
  for (int j = 0; j < 8; ++j) a[j] = src[m * ld + kb + j];
#pragma unroll
  for (int j = 0; j < 8; ++j) a[8 + j] = src[m * ld + 16 + kb + j];
  return a;
}

// ---------------------------------------------------------------------------
// f32 -> f16 conversion
// ---------------------------------------------------------------------------
__global__ void f32_to_f16_kernel(const float* __restrict__ in,
                                  _Float16* __restrict__ out, int n) {
  int i = blockIdx.x * blockDim.x + threadIdx.x;
  if (i < n) out[i] = (_Float16)in[i];
}

// ---------------------------------------------------------------------------
// Tiled WMMA GEMM: C[M][N] = A[M][K](f16) * B[K][N](f16) + bias[N], f32 out.
// Block tile 128x128, 8 waves (4Mx2N), wave tile 32x64 = 2x4 WMMA tiles,
// K-step 32. A tile staged with gfx1250 async global->LDS DMA when available;
// B tile staged TRANSPOSED in LDS so both A and B fragments are contiguous
// 8-half runs (ds_load_b128, no scalar repack).
// ---------------------------------------------------------------------------
__global__ __launch_bounds__(256)
void gemm_f16_wmma(const _Float16* __restrict__ A, const _Float16* __restrict__ B,
                   const float* __restrict__ bias, float* __restrict__ C,
                   int M, int N, int K) {
  __shared__ _Float16 As[128][40];   // [m][k], 80B row stride (16B aligned)
  __shared__ _Float16 BsT[128][40];  // [n][k], transposed stage
  const int t    = threadIdx.x;
  const int wave = t >> 5;
  const int wm   = wave >> 1;        // 0..3 -> 32 rows each
  const int wn   = wave & 1;         // 0..1 -> 64 cols each
  const int m0   = blockIdx.y * 128;
  const int n0   = blockIdx.x * 128;

  v8f acc[2][4] = {};

  for (int k0 = 0; k0 < K; k0 += 32) {
    __syncthreads();
    // A tile 128x32 halves: 256 threads x 2 x 16B
#pragma unroll
    for (int it = 0; it < 2; ++it) {
      const int li = (t + it * 256) * 8;
      const int r = li >> 5, c = li & 31;
      const _Float16* gsrc = A + (size_t)(m0 + r) * K + k0 + c;
#if HAVE_ASYNC_LDS
      __builtin_amdgcn_global_load_async_to_lds_b128(
          (gptr_v4i)gsrc, (lptr_v4i)&As[r][c], 0, 0);
#else
      *(uint4*)&As[r][c] = *(const uint4*)gsrc;
#endif
    }
    // B tile 32(k) x 128(n) halves: vector load along n, transpose into BsT
#pragma unroll
    for (int it = 0; it < 2; ++it) {
      const int li = (t + it * 256) * 8;
      const int r = li >> 7, c = li & 127;   // r = k row, c = n col
      union { uint4 u; _Float16 h[8]; } tmp;
      tmp.u = *(const uint4*)(B + (size_t)(k0 + r) * N + n0 + c);
#pragma unroll
      for (int j = 0; j < 8; ++j) BsT[c + j][r] = tmp.h[j];
    }
#if HAVE_ASYNC_LDS
    wait_asynccnt0();
#endif
    __syncthreads();

    v16h af[2], bf[4];
#pragma unroll
    for (int tm = 0; tm < 2; ++tm) af[tm] = frag_kmaj(&As[wm * 32 + tm * 16][0], 40);
#pragma unroll
    for (int tn = 0; tn < 4; ++tn) bf[tn] = frag_kmaj(&BsT[wn * 64 + tn * 16][0], 40);
#pragma unroll
    for (int tm = 0; tm < 2; ++tm)
#pragma unroll
      for (int tn = 0; tn < 4; ++tn)
        acc[tm][tn] = __builtin_amdgcn_wmma_f32_16x16x32_f16(
            false, af[tm], false, bf[tn], (short)0, acc[tm][tn], false, false);
  }

  const int lane = t & 31, nl = lane & 15, hf = lane >> 4;
#pragma unroll
  for (int tm = 0; tm < 2; ++tm)
#pragma unroll
    for (int tn = 0; tn < 4; ++tn)
#pragma unroll
      for (int r = 0; r < 8; ++r) {
        const int row = m0 + wm * 32 + tm * 16 + r + hf * 8;
        const int col = n0 + wn * 64 + tn * 16 + nl;
        C[(size_t)row * N + col] = acc[tm][tn][r] + bias[col];
      }
}

// ---------------------------------------------------------------------------
// RoPE + repack: qkv f32 [S][3E] -> q/k f16 [H][S][D] (K-contiguous for QK^T)
// and v f16 [H][D][S] (key-contiguous for PV). Q gets the D^-0.5 scale.
// ---------------------------------------------------------------------------
__global__ void rope_pack_kernel(const float* __restrict__ qkv,
                                 _Float16* __restrict__ qh,
                                 _Float16* __restrict__ kh,
                                 _Float16* __restrict__ vt) {
  const int idx = blockIdx.x * blockDim.x + threadIdx.x;  // 0..S*E-1
  const int s = idx >> 10;
  const int e = idx & 1023;
  const int h = e >> 6;
  const int d = e & 63;
  const float* row = qkv + (size_t)s * (3 * EMB);
  const float q = row[e];
  const float k = row[EMB + e];
  const float v = row[2 * EMB + e];
  const int ep = (d < 32) ? (e + 32) : (e - 32);
  const float q2 = row[ep];
  const float k2 = row[EMB + ep];
  const int i = d & 31;
  const float invf = powf(10000.0f, -(float)(2 * i) / 64.0f);
  const float ang = (float)s * invf;
  const float c = cosf(ang), sn = sinf(ang);
  const float qr = (d < 32) ? (q * c - q2 * sn) : (q * c + q2 * sn);
  const float kr = (d < 32) ? (k * c - k2 * sn) : (k * c + k2 * sn);
  const size_t o = (size_t)h * S_LEN * HDIM + (size_t)s * HDIM + d;
  qh[o] = (_Float16)(qr * 0.125f);
  kh[o] = (_Float16)kr;
  vt[(size_t)h * HDIM * S_LEN + (size_t)d * S_LEN + s] = (_Float16)v;
}

// ---------------------------------------------------------------------------
// Causal flash attention: one wave32 per (16 query rows, head).
// QK^T: 4 WMMAs per 32-key step (D=64 = 2 K-chunks); online softmax in f32;
// P re-laid out through LDS into an A fragment; PV: 4 WMMAs per step using
// key-contiguous V^T (vectorized global b128 fragment loads). Next key block
// prefetched (global_prefetch_b8) to overlap softmax VALU with memory.
// ---------------------------------------------------------------------------
__global__ __launch_bounds__(32)
void flash_attn_kernel(const _Float16* __restrict__ qh,
                       const _Float16* __restrict__ kh,
                       const _Float16* __restrict__ vt,
                       _Float16* __restrict__ ctx) {
  __shared__ _Float16 pl[16][32];
  const int qb = blockIdx.x * 16;
  const int h  = blockIdx.y;
  const int lane = threadIdx.x & 31, nl = lane & 15, hf = lane >> 4;
  const _Float16* Q  = qh + (size_t)h * S_LEN * HDIM + (size_t)qb * HDIM;
  const _Float16* Kh = kh + (size_t)h * S_LEN * HDIM;
  const _Float16* Vt = vt + (size_t)h * HDIM * S_LEN;

  v16h qa0 = frag_kmaj(Q, HDIM);
  v16h qa1 = frag_kmaj(Q + 32, HDIM);

  v8f o[4] = {};
  float mrow[8], lrow[8];
#pragma unroll
  for (int r = 0; r < 8; ++r) { mrow[r] = -1e30f; lrow[r] = 0.0f; }

  const int nkb = (qb + 16 + 31) >> 5;  // causal: keys <= qb+15
  for (int kb = 0; kb < nkb; ++kb) {
    const int kbase = kb << 5;

    // prefetch next key/value block while this one is consumed
    if (kb + 1 < nkb) {
      __builtin_prefetch((const void*)(Kh + (size_t)(kbase + 32) * HDIM + lane * 4), 0, 1);
      __builtin_prefetch((const void*)(Vt + (size_t)(lane * 2) * S_LEN + kbase + 32), 0, 1);
    }

    v8f s0 = {}, s1 = {};
    {
      // B fragment of K^T: element (kdim, key) = Kh[key][kdim], key-major rows
      v16h b00 = frag_kmaj(Kh + (size_t)kbase * HDIM, HDIM);
      v16h b01 = frag_kmaj(Kh + (size_t)kbase * HDIM + 32, HDIM);
      s0 = __builtin_amdgcn_wmma_f32_16x16x32_f16(false, qa0, false, b00, (short)0, s0, false, false);
      s0 = __builtin_amdgcn_wmma_f32_16x16x32_f16(false, qa1, false, b01, (short)0, s0, false, false);
      v16h b10 = frag_kmaj(Kh + (size_t)(kbase + 16) * HDIM, HDIM);
      v16h b11 = frag_kmaj(Kh + (size_t)(kbase + 16) * HDIM + 32, HDIM);
      s1 = __builtin_amdgcn_wmma_f32_16x16x32_f16(false, qa0, false, b10, (short)0, s1, false, false);
      s1 = __builtin_amdgcn_wmma_f32_16x16x32_f16(false, qa1, false, b11, (short)0, s1, false, false);
    }

    float p0[8], p1[8];
#pragma unroll
    for (int r = 0; r < 8; ++r) {
      const int qrow = qb + r + hf * 8;
      const float v0 = (kbase + nl      > qrow) ? -1e30f : s0[r];
      const float v1 = (kbase + 16 + nl > qrow) ? -1e30f : s1[r];
      float rm = fmaxf(v0, v1);
      rm = fmaxf(rm, __shfl_xor(rm, 1));
      rm = fmaxf(rm, __shfl_xor(rm, 2));
      rm = fmaxf(rm, __shfl_xor(rm, 4));
      rm = fmaxf(rm, __shfl_xor(rm, 8));
      const float mnew = fmaxf(mrow[r], rm);
      const float corr = __expf(mrow[r] - mnew);
      const float e0 = __expf(v0 - mnew);
      const float e1 = __expf(v1 - mnew);
      float ps = e0 + e1;
      ps += __shfl_xor(ps, 1);
      ps += __shfl_xor(ps, 2);
      ps += __shfl_xor(ps, 4);
      ps += __shfl_xor(ps, 8);
      lrow[r] = lrow[r] * corr + ps;
      mrow[r] = mnew;
      p0[r] = e0; p1[r] = e1;
#pragma unroll
      for (int ct = 0; ct < 4; ++ct) o[ct][r] = o[ct][r] * corr;
    }

    __syncthreads();  // previous iteration's pl reads complete
#pragma unroll
    for (int r = 0; r < 8; ++r) {
      const int m = r + hf * 8;
      pl[m][nl]      = (_Float16)p0[r];
      pl[m][16 + nl] = (_Float16)p1[r];
    }
    __syncthreads();

    v16h pa = frag_kmaj(&pl[0][0], 32);
#pragma unroll
    for (int ct = 0; ct < 4; ++ct) {
      // B fragment of V: element (key, d) = Vt[d][key], d-major rows
      v16h bv = frag_kmaj(Vt + (size_t)(ct * 16) * S_LEN + kbase, S_LEN);
      o[ct] = __builtin_amdgcn_wmma_f32_16x16x32_f16(false, pa, false, bv, (short)0, o[ct], false, false);
    }
  }

#pragma unroll
  for (int ct = 0; ct < 4; ++ct)
#pragma unroll
    for (int r = 0; r < 8; ++r) {
      const int row = qb + r + hf * 8;
      const int col = h * HDIM + ct * 16 + nl;
      ctx[(size_t)row * EMB + col] = (_Float16)(o[ct][r] / lrow[r]);
    }
}

// ---------------------------------------------------------------------------
// Host-side launch
// ---------------------------------------------------------------------------
extern "C" void kernel_launch(void* const* d_in, const int* in_sizes, int n_in,
                              void* d_out, int out_size, void* d_ws, size_t ws_size,
                              hipStream_t stream) {
  const float* x      = (const float*)d_in[0];
  // d_in[1] = mask (causal; enforced analytically in the kernel)
  const float* wqkv_w = (const float*)d_in[2];
  const float* wqkv_b = (const float*)d_in[3];
  const float* out_w  = (const float*)d_in[4];
  const float* out_b  = (const float*)d_in[5];
  float* out = (float*)d_out;

  char* ws = (char*)d_ws;
  // workspace carve (bytes)
  _Float16* xh     = (_Float16*)(ws + 0);            //  8 MB: 4096x1024 f16
  _Float16* wqkvh  = (_Float16*)(ws + 8388608);      //  6 MB: 1024x3072 f16
  _Float16* outwh  = (_Float16*)(ws + 14680064);     //  2 MB: 1024x1024 f16
  float*    qkv    = (float*)   (ws + 16777216);     // 48 MB: 4096x3072 f32
  _Float16* qh     = (_Float16*)(ws + 67108864);     //  8 MB: [H][S][D] f16
  _Float16* kh     = (_Float16*)(ws + 75497472);     //  8 MB: [H][S][D] f16
  _Float16* vt     = (_Float16*)(ws + 83886080);     //  8 MB: [H][D][S] f16
  _Float16* ctxh   = (_Float16*)(ws + 92274688);     //  8 MB: 4096x1024 f16

  // 1) f32 -> f16 conversions
  {
    int n = S_LEN * EMB;
    f32_to_f16_kernel<<<(n + 255) / 256, 256, 0, stream>>>(x, xh, n);
    n = EMB * 3 * EMB;
    f32_to_f16_kernel<<<(n + 255) / 256, 256, 0, stream>>>(wqkv_w, wqkvh, n);
    n = EMB * EMB;
    f32_to_f16_kernel<<<(n + 255) / 256, 256, 0, stream>>>(out_w, outwh, n);
  }

  // 2) QKV projection: [4096x1024] x [1024x3072] + bias -> qkv f32
  {
    dim3 grid((3 * EMB) / 128, S_LEN / 128);
    gemm_f16_wmma<<<grid, 256, 0, stream>>>(xh, wqkvh, wqkv_b, qkv,
                                            S_LEN, 3 * EMB, EMB);
  }

  // 3) RoPE + per-head f16 repack (Q scaled by D^-0.5; V transposed)
  {
    const int n = S_LEN * EMB;
    rope_pack_kernel<<<n / 256, 256, 0, stream>>>(qkv, qh, kh, vt);
  }

  // 4) Causal flash attention
  {
    dim3 grid(S_LEN / 16, NHEAD);
    flash_attn_kernel<<<grid, 32, 0, stream>>>(qh, kh, vt, ctxh);
  }

  // 5) Output projection: [4096x1024] x [1024x1024] + bias -> d_out f32
  {
    dim3 grid(EMB / 128, S_LEN / 128);
    gemm_f16_wmma<<<grid, 256, 0, stream>>>(ctxh, outwh, out_b, out,
                                            S_LEN, EMB, EMB);
  }
}
